// PredictionEncoder_86079734546933
// MI455X (gfx1250) — matrix operand order repaired and checked by reading, not compiled, not run on hardware
//
#include <hip/hip_runtime.h>
#include <hip/hip_bf16.h>
#include <math.h>

// ---------------------------------------------------------------------------
// Types / WMMA helpers (CDNA5 gfx1250, wave32)
// ---------------------------------------------------------------------------
typedef _Float16 h16;
typedef __attribute__((ext_vector_type(16))) _Float16 v16h;
typedef __attribute__((ext_vector_type(8)))  float    v8f;
typedef __attribute__((ext_vector_type(4)))  unsigned u32x4;
typedef __attribute__((ext_vector_type(8)))  int      i32x8;
typedef __attribute__((ext_vector_type(4)))  int      i32x4;
typedef int v4i __attribute__((vector_size(16)));   // matches builtin proto V4i

#define DEV static __device__ __forceinline__

// ---- CDNA5 async / TDM availability probes --------------------------------
#if defined(__has_builtin)
#  if __has_builtin(__builtin_amdgcn_global_load_async_to_lds_b128)
#    define HAVE_ASYNC_LDS 1
#  endif
#  if __has_builtin(__builtin_amdgcn_tensor_load_to_lds) && __has_builtin(__builtin_amdgcn_s_wait_tensorcnt)
#    define HAVE_TDM 1
#  endif
#endif

// async 16B global->LDS copy (falls back to plain copy)
// builtin proto: (v4i addrspace(1)*, v4i addrspace(3)*, imm offset, imm cpol)
DEV void async_g2l_b128(void* lds, const void* g) {
#ifdef HAVE_ASYNC_LDS
  __builtin_amdgcn_global_load_async_to_lds_b128(
      (__attribute__((address_space(1))) v4i*)(void*)g,
      (__attribute__((address_space(3))) v4i*)lds, 0, 0);
#else
  __builtin_memcpy(lds, g, 16);
#endif
}

DEV void wait_async0() {
#ifdef HAVE_ASYNC_LDS
#  if __has_builtin(__builtin_amdgcn_s_wait_asynccnt)
  __builtin_amdgcn_s_wait_asynccnt(0);
#  else
  asm volatile("s_wait_asynccnt 0x0" ::: "memory");
#  endif
#endif
}

// Tensor Data Mover: 1-D tile copy global -> LDS (nbytes multiple of 4).
// D# built per CDNA5 ISA (data_size=4B, 1-D tile).
DEV void tdm_load_1d(void* lds, const void* g, unsigned nbytes) {
#ifdef HAVE_TDM
  unsigned elems = nbytes >> 2;
  auto p3 = (__attribute__((address_space(3))) char*)lds;
  unsigned laddr = (unsigned)(size_t)p3;
  unsigned long long ga = (unsigned long long)(size_t)g;
  u32x4 g0 = { 1u,                                  // count=1, user desc
               laddr,                               // lds_addr (bytes)
               (unsigned)ga,                        // global_addr[31:0]
               (unsigned)((ga >> 32) & 0x01FFFFFFu) | (2u << 30) }; // [56:32] | type=2
  i32x8 g1 = { (int)(2u << 16),                     // data_size=2 (4 bytes)
               (int)((elems & 0xFFFFu) << 16),      // tensor_dim0[15:0]
               (int)(((elems >> 16) & 0xFFFFu) | (1u << 16)), // dim0 hi | tensor_dim1=1
               (int)((elems & 0xFFFFu) << 16),      // tile_dim0 = elems
               0,                                   // tile_dim1=0 (unused), tile_dim2=0
               (int)elems,                          // tensor_dim0_stride lo
               0, 0 };
  i32x4 z4 = { 0, 0, 0, 0 };
#  if __clang_major__ >= 23
  i32x8 z8 = { 0, 0, 0, 0, 0, 0, 0, 0 };
  __builtin_amdgcn_tensor_load_to_lds(g0, g1, z4, z4, z8, 0);
#  else
  __builtin_amdgcn_tensor_load_to_lds(g0, g1, z4, z4, 0);
#  endif
#else
  (void)lds; (void)g; (void)nbytes;
#endif
}

DEV void tdm_wait() {
#ifdef HAVE_TDM
  __builtin_amdgcn_s_wait_tensorcnt((short)0);
#endif
}

// A fragment: 16x32 f16, row-major A with leading dim lda.
DEV v16h load_a_frag(const h16* A, int lda, int m0, int k0, int lane) {
  const h16* p = A + (size_t)(m0 + (lane & 15)) * lda + (size_t)(k0 + ((lane >> 4) << 3));
  v16h r;
  __builtin_memcpy(&r, p, 16);
  __builtin_memcpy(((char*)&r) + 16, p + 16, 16);
  return r;
}

// B fragment: 32x16 f16 where B[k][n] = W[n][k]; W is (N x K) row-major.
DEV v16h load_bt_frag(const h16* W, int ldw, int n0, int k0, int lane) {
  const h16* p = W + (size_t)(n0 + (lane & 15)) * ldw + (size_t)(k0 + ((lane >> 4) << 3));
  v16h r;
  __builtin_memcpy(&r, p, 16);
  __builtin_memcpy(((char*)&r) + 16, p + 16, 16);
  return r;
}

DEV v8f wmma_f16(v16h a, v16h b, v8f c) {
  return __builtin_amdgcn_wmma_f32_16x16x32_f16(false, a, false, b, (short)0, c, false, false);
}

DEV float sigmf(float x) { return 1.0f / (1.0f + __expf(-x)); }

// splitmix64 -> uniform(0,1)
DEV float u01(unsigned long long x) {
  x += 0x9E3779B97F4A7C15ull;
  x = (x ^ (x >> 30)) * 0xBF58476D1CE4E5B9ull;
  x = (x ^ (x >> 27)) * 0x94D049BB133111EBull;
  x ^= x >> 31;
  return ((float)(unsigned)(x >> 40) + 0.5f) * (1.0f / 16777216.0f);
}

// ---------------------------------------------------------------------------
// Problem constants
// ---------------------------------------------------------------------------
#define BB   256
#define NN   32
#define BN   8192      // B*N
#define DIN  64
#define HH   128
#define AA   128
#define NM1  31        // N-1
#define TAUI 100.0f    // 1/TAU

// ---------------------------------------------------------------------------
// Generic tiled WMMA GEMM:  C[MxN] = act(A[MxK] @ W[NxK]^T + bias)
// Each wave computes a (16*TM) x (16*TN) tile: TMxTN WMMA accumulators,
// A-frags reused TN times, B-frags TM times (1.5 loads/WMMA at 2x4).
// ---------------------------------------------------------------------------
template <int TM, int TN>
__global__ void __launch_bounds__(128)
gemm16_kernel(const h16* __restrict__ A, int lda,
              const h16* __restrict__ W, int ldw,
              const float* __restrict__ bias,
              float* __restrict__ C, h16* __restrict__ C16,
              int M, int N, int K, int act)
{
  int wave = threadIdx.x >> 5;
  int lane = threadIdx.x & 31;
  int bm = M / (16 * TM);
  int bn = N / (16 * TN);
  int tile = blockIdx.x * 4 + wave;
  if (tile >= bm * bn) return;
  int tm = tile / bn, tn = tile - tm * bn;
  int m0 = tm * 16 * TM, n0 = tn * 16 * TN;

  v8f acc[TM][TN] = {};
  for (int k0 = 0; k0 < K; k0 += 32) {
    v16h a[TM], b[TN];
#pragma unroll
    for (int i = 0; i < TM; ++i) a[i] = load_a_frag(A, lda, m0 + 16 * i, k0, lane);
#pragma unroll
    for (int j = 0; j < TN; ++j) b[j] = load_bt_frag(W, ldw, n0 + 16 * j, k0, lane);
#pragma unroll
    for (int i = 0; i < TM; ++i)
#pragma unroll
      for (int j = 0; j < TN; ++j)
        acc[i][j] = wmma_f16(a[i], b[j], acc[i][j]);
  }

  int nl = lane & 15, mb = (lane >> 4) << 3;
#pragma unroll
  for (int i = 0; i < TM; ++i) {
#pragma unroll
    for (int j = 0; j < TN; ++j) {
      int n = n0 + 16 * j + nl;
      float bv = bias ? bias[n] : 0.0f;
#pragma unroll
      for (int v = 0; v < 8; ++v) {
        float x = acc[i][j][v] + bv;
        if (act == 1) x = fmaxf(x, 0.0f);
        size_t off = (size_t)(m0 + 16 * i + mb + v) * N + n;
        if (C)   C[off]   = x;
        if (C16) C16[off] = (h16)x;
      }
    }
  }
}

// ---------------------------------------------------------------------------
// f32 -> f16 conversion with optional column slice + zero pad
// ---------------------------------------------------------------------------
__global__ void cvt16_kernel(const float* __restrict__ src, int srcLd, int srcOff, int srcCols,
                             h16* __restrict__ dst, int dstLd, int rows)
{
  int idx = blockIdx.x * blockDim.x + threadIdx.x;
  int total = rows * dstLd;
  if (idx >= total) return;
  int r = idx / dstLd, c = idx - r * dstLd;
  float v = (c < srcCols) ? src[(size_t)r * srcLd + srcOff + c] : 0.0f;
  dst[idx] = (h16)v;
}

// input_gat = [obs[:43], latent[16]] -> 8192 x 64 f16 (cols 59..63 zero)
__global__ void build_gat_kernel(const float* __restrict__ inp, h16* __restrict__ X)
{
  int idx = blockIdx.x * blockDim.x + threadIdx.x;
  if (idx >= BN * 64) return;
  int r = idx >> 6, c = idx & 63;
  float v = 0.0f;
  if (c < 43)      v = inp[(size_t)r * DIN + c];
  else if (c < 59) v = inp[(size_t)r * DIN + 48 + (c - 43)];
  X[idx] = (h16)v;
}

// m = relu(input_motion @ mot_W^T + mot_b), input_motion = inp[:,43:47]
__global__ void motion_kernel(const float* __restrict__ inp, const float* __restrict__ motW,
                              const float* __restrict__ motb, h16* __restrict__ m16)
{
  int idx = blockIdx.x * blockDim.x + threadIdx.x;
  if (idx >= BN * HH) return;
  int r = idx >> 7, c = idx & 127;
  const float* xi = inp + (size_t)r * DIN + 43;
  float s = motb[c];
#pragma unroll
  for (int j = 0; j < 4; ++j) s += xi[j] * motW[c * 4 + j];
  m16[idx] = (h16)fmaxf(s, 0.0f);
}

// ---------------------------------------------------------------------------
// Persistent bi-GRU: one block = 16 rows for all 31 steps.
// Whh (96KB f16) staged into LDS once via TDM; GIa tile (24KB) staged via
// async global->LDS; GIb gather rows prefetched one step ahead.
// Hard-gate logits fused into the epilogue.
// LDS carve (dynamic, 156KB/WG out of the 320KB WGP pool):
//   sWhh 98304 | sGIa 24576 | sGh 24576 | sH 8192 | sH16 4096 | sL 128
// ---------------------------------------------------------------------------
#define GRU_SMEM (98304 + 24576 + 24576 + 8192 + 4096 + 128)

__global__ void __launch_bounds__(128)
gru_seq_kernel(const float* __restrict__ GIa, const float* __restrict__ GIb,
               const float* __restrict__ bhh, const h16* __restrict__ Whh,
               const float* __restrict__ hardW, const float* __restrict__ hardb,
               float* __restrict__ LfOut, const float* __restrict__ LfIn,
               float* __restrict__ hw, int reverse)
{
  extern __shared__ char smem[];
  h16*   sWhh = (h16*)smem;                          // 384x128 f16
  float* sGIa = (float*)(smem + 98304);              // 16x384
  float* sGh  = (float*)(smem + 98304 + 24576);      // 16x384
  float* sH   = (float*)(smem + 98304 + 2 * 24576);  // 16x128
  h16*   sH16 = (h16*)(smem + 98304 + 2 * 24576 + 8192);
  float* sL   = (float*)(smem + 98304 + 2 * 24576 + 8192 + 4096);

  int tid = threadIdx.x;
  int wave = tid >> 5, lane = tid & 31;
  int row0 = blockIdx.x << 4;

  // ---- stage Whh via Tensor Data Mover (one wave issues; EXEC ignored) ----
#ifdef HAVE_TDM
  if (tid < 32) tdm_load_1d(sWhh, Whh, 384 * 128 * 2);
#else
  for (int e = tid * 8; e < 384 * 128; e += 128 * 8) async_g2l_b128(sWhh + e, Whh + e);
#endif
  // ---- stage this block's GIa rows (contiguous 24KB) via async-to-LDS ----
  for (int e = tid * 4; e < 16 * 384; e += 128 * 4)
    async_g2l_b128(sGIa + e, GIa + (size_t)row0 * 384 + e);

  for (int e = tid; e < 16 * 128; e += 128) { sH[e] = 0.0f; sH16[e] = (h16)0.0f; }
  tdm_wait();
  wait_async0();
  __syncthreads();

  for (int s = 0; s < NM1; ++s) {
    int t = reverse ? (NM1 - 1 - s) : s;

    // ---- phase 1: gh = h @ Whh^T (16x384) from LDS-resident h and Whh ----
#pragma unroll
    for (int q = 0; q < 6; ++q) {
      int tn = wave * 6 + q;
      v8f acc = {};
#pragma unroll
      for (int kk = 0; kk < 4; ++kk) {
        v16h a = load_a_frag((const h16*)sH16, 128, 0, kk * 32, lane);
        v16h b = load_bt_frag(sWhh, 128, tn * 16, kk * 32, lane);
        acc = wmma_f16(a, b, acc);
      }
      int n = tn * 16 + (lane & 15);
      int mb = (lane >> 4) << 3;
#pragma unroll
      for (int v = 0; v < 8; ++v) sGh[(mb + v) * 384 + n] = acc[v];
    }
    __syncthreads();

    // ---- phase 2: gates, update h ----
    for (int e = tid; e < 16 * 128; e += 128) {
      int m = e >> 7, c = e & 127;
      int row = row0 + m;
      int b = row >> 5, i = row & 31;
      int jj = (t < i) ? t : t + 1;
      int rowB = (b << 5) + jj;
      const float* ga = sGIa + m * 384;
      const float* gb = GIb + (size_t)rowB * 384;
      float ghr = sGh[m * 384 + c]       + bhh[c];
      float ghz = sGh[m * 384 + c + 128] + bhh[c + 128];
      float ghn = sGh[m * 384 + c + 256] + bhh[c + 256];
      float r = sigmf(ga[c]       + gb[c]       + ghr);
      float z = sigmf(ga[c + 128] + gb[c + 128] + ghz);
      float n = tanhf(ga[c + 256] + gb[c + 256] + r * ghn);
      float hp = sH[e];
      float hn = (1.0f - z) * n + z * hp;
      sH[e] = hn;
      sH16[e] = (h16)hn;
    }
    __syncthreads();

    // ---- epilogue: fused hard-gate logits (2 per row) + GIb prefetch ----
    if (tid < 32) {
      int m = tid & 15, g = tid >> 4;
      const float* wrow = hardW + g * 256 + (reverse ? 128 : 0);
      float d = 0.0f;
      for (int h = 0; h < 128; ++h) d += sH[m * 128 + h] * wrow[h];
      int row = row0 + m;
      if (!reverse) LfOut[((size_t)t * BN + row) * 2 + g] = d;
      else          sL[m * 2 + g] = d + hardb[g] + LfIn[((size_t)t * BN + row) * 2 + g];
    } else if (tid >= 64 && tid < 80 && s + 1 < NM1) {
      // prefetch next step's gather rows of GIb
      int tnx = reverse ? (NM1 - 2 - s) : (s + 1);
      int row = row0 + (tid - 64);
      int b = row >> 5, i = row & 31;
      int jj = (tnx < i) ? tnx : tnx + 1;
      const float* pb = GIb + (size_t)((b << 5) + jj) * 384;
      __builtin_prefetch(pb, 0, 1);
      __builtin_prefetch(pb + 256, 0, 1);
      __builtin_prefetch(pb + 512, 0, 1);
    }
    __syncthreads();
    if (reverse && tid < 16) {
      int m = tid, row = row0 + m;
      unsigned long long rl = (unsigned long long)((size_t)row * NM1 + t);
      float l0 = sL[m * 2 + 0], l1 = sL[m * 2 + 1];
      float g0 = -logf(-logf(u01(rl * 2 + 0)));
      float g1 = -logf(-logf(u01(rl * 2 + 1)));
      float d = ((l1 + g1) - (l0 + g0)) * TAUI;
      hw[(size_t)row * NM1 + t] = 1.0f / (1.0f + __expf(-d));
    }
    __syncthreads();
  }
}

// ---------------------------------------------------------------------------
// Masked softmax attention: x[row] = sum_j softmax(q.k_j/sqrt(A)) * hw * v_j
// ---------------------------------------------------------------------------
__global__ void __launch_bounds__(128)
attention_kernel(const float* __restrict__ q, const float* __restrict__ k,
                 const float* __restrict__ v, const float* __restrict__ hw,
                 h16* __restrict__ x16)
{
  __shared__ float sQ[128];
  __shared__ float sS[32];
  __shared__ float sW[32];
  int row = blockIdx.x, tid = threadIdx.x;
  int b = row >> 5, i = row & 31;
  sQ[tid] = q[(size_t)row * AA + tid];
  __syncthreads();
  if (tid < NM1) {
    int jj = (tid < i) ? tid : tid + 1;
    const float* kr = k + (size_t)((b << 5) + jj) * AA;
    float d = 0.0f;
    for (int c = 0; c < AA; ++c) d += sQ[c] * kr[c];
    sS[tid] = d * 0.088388347648318440550f; // 1/sqrt(128)
  }
  __syncthreads();
  if (tid < NM1) {
    float mx = -1e30f;
    for (int j = 0; j < NM1; ++j) mx = fmaxf(mx, sS[j]);
    float se = 0.0f;
    for (int j = 0; j < NM1; ++j) se += __expf(sS[j] - mx);
    sW[tid] = (__expf(sS[tid] - mx) / se) * hw[(size_t)row * NM1 + tid];
  }
  __syncthreads();
  float acc = 0.0f;
  for (int j = 0; j < NM1; ++j) {
    int jj = (j < i) ? j : j + 1;
    acc += sW[j] * v[(size_t)((b << 5) + jj) * AA + tid];
  }
  x16[(size_t)row * AA + tid] = (h16)acc;
}

// ---------------------------------------------------------------------------
// GRU cell elementwise (gi/gh already have biases)
// ---------------------------------------------------------------------------
__global__ void grucell_kernel(const float* __restrict__ gi, const float* __restrict__ gh,
                               const float* __restrict__ hidden, int hoff,
                               float* __restrict__ hcF, h16* __restrict__ hc16, int coff)
{
  int idx = blockIdx.x * blockDim.x + threadIdx.x;
  if (idx >= BN * HH) return;
  int row = idx >> 7, c = idx & 127;
  const float* a = gi + (size_t)row * 384;
  const float* g = gh + (size_t)row * 384;
  float r = sigmf(a[c] + g[c]);
  float z = sigmf(a[c + 128] + g[c + 128]);
  float n = tanhf(a[c + 256] + r * g[c + 256]);
  float hp = hidden[(size_t)row * 256 + hoff + c];
  float hn = (1.0f - z) * n + z * hp;
  hcF[(size_t)row * 256 + coff + c] = hn;
  hc16[(size_t)row * 256 + coff + c] = (h16)hn;
}

// ---------------------------------------------------------------------------
// BatchNorm column stats + apply
// ---------------------------------------------------------------------------
__global__ void __launch_bounds__(256)
colstats_kernel(const float* __restrict__ X, int M, int N, float* __restrict__ stats)
{
  __shared__ float s1[256], s2[256];
  int n = blockIdx.x, tid = threadIdx.x;
  float a = 0.0f, b = 0.0f;
  for (int r = tid; r < M; r += 256) {
    float x = X[(size_t)r * N + n];
    a += x; b += x * x;
  }
  s1[tid] = a; s2[tid] = b;
  __syncthreads();
  for (int st = 128; st > 0; st >>= 1) {
    if (tid < st) { s1[tid] += s1[tid + st]; s2[tid] += s2[tid + st]; }
    __syncthreads();
  }
  if (tid == 0) {
    float mu = s1[0] / (float)M;
    float var = s2[0] / (float)M - mu * mu;
    stats[2 * n]     = mu;
    stats[2 * n + 1] = rsqrtf(var + 1e-5f);
  }
}

__global__ void bnrelu_kernel(const float* __restrict__ X, const float* __restrict__ stats,
                              const float* __restrict__ g, const float* __restrict__ be,
                              int M, int N, float* __restrict__ Y, h16* __restrict__ Y16)
{
  int idx = blockIdx.x * blockDim.x + threadIdx.x;
  if (idx >= M * N) return;
  int c = idx % N;
  float y = (X[idx] - stats[2 * c]) * stats[2 * c + 1] * g[c] + be[c];
  y = fmaxf(y, 0.0f);
  if (Y)   Y[idx]   = y;
  if (Y16) Y16[idx] = (h16)y;
}

// wh = 0.05 * [pe, hp]   (aw == 1 exactly: softmax over a length-1 axis)
__global__ void buildwh_kernel(const float* __restrict__ inp, const float* __restrict__ spatW,
                               const float* __restrict__ spatb, const float* __restrict__ hp,
                               h16* __restrict__ wh)
{
  int idx = blockIdx.x * blockDim.x + threadIdx.x;
  if (idx >= BN * 256) return;
  int row = idx >> 8, c = idx & 255;
  float v;
  if (c < 128) {
    const float* xi = inp + (size_t)row * DIN;
    v = xi[1] * spatW[c * 2] + xi[2] * spatW[c * 2 + 1] + spatb[c];
  } else {
    v = hp[(size_t)row * 128 + (c - 128)];
  }
  wh[idx] = (h16)(0.05f * v);
}

// ---------------------------------------------------------------------------
// Host orchestration
// ---------------------------------------------------------------------------
enum {
  IN_INP = 0, IN_HIDDEN,
  P_ENC_W, P_ENC_B, P_GF_WIH, P_GF_WHH, P_GF_BIH, P_GF_BHH,
  P_GB_WIH, P_GB_WHH, P_GB_BIH, P_GB_BHH, P_HARD_W, P_HARD_B,
  P_Q_W, P_K_W, P_V_W, P_V_B,
  P_GC_WIH, P_GC_WHH, P_GC_BIH, P_GC_BHH, P_MOT_W, P_MOT_B,
  P_MC_WIH, P_MC_WHH, P_MC_BIH, P_MC_BHH,
  P_MLP_W1, P_MLP_B1, P_MLP_G1, P_MLP_BE1, P_MLP_W2, P_MLP_B2, P_MLP_G2, P_MLP_BE2,
  P_SPAT_W, P_SPAT_B, P_VEL_W, P_VEL_B,
  P_AT_W1, P_AT_B1, P_AT_G1, P_AT_BE1, P_AT_W2, P_AT_B2, P_AT_G2, P_AT_BE2,
  P_PP_W1, P_PP_B1, P_PP_G1, P_PP_BE1, P_PP_W2, P_PP_B2, P_PP_G2, P_PP_BE2
};

extern "C" void kernel_launch(void* const* d_in, const int* in_sizes, int n_in,
                              void* d_out, int out_size, void* d_ws, size_t ws_size,
                              hipStream_t stream)
{
  (void)in_sizes; (void)n_in; (void)out_size; (void)ws_size;
  const float* inp    = (const float*)d_in[IN_INP];
  const float* hidden = (const float*)d_in[IN_HIDDEN];
  auto P = [&](int i) { return (const float*)d_in[i]; };

  float* out_pool = (float*)d_out;              // 8192 x 32
  float* out_hc   = out_pool + BN * 32;         // 8192 x 256

  // ---- workspace carve ----
  char* base = (char*)d_ws;
  size_t cur = 0;
  auto allocB = [&](size_t bytes) -> char* {
    char* p = base + cur;
    cur = (cur + bytes + 255) & ~(size_t)255;
    return p;
  };
  auto allocH = [&](size_t n) { return (h16*)allocB(n * sizeof(h16)); };
  auto allocF = [&](size_t n) { return (float*)allocB(n * sizeof(float)); };

  h16* Xgat    = allocH((size_t)BN * 64);
  h16* hEnc16  = allocH((size_t)BN * HH);
  h16* encW16  = allocH(128 * 64);
  h16* gfA16   = allocH(384 * 128);
  h16* gfB16   = allocH(384 * 128);
  h16* gbA16   = allocH(384 * 128);
  h16* gbB16   = allocH(384 * 128);
  h16* gfWhh16 = allocH(384 * 128);
  h16* gbWhh16 = allocH(384 * 128);
  h16* qW16    = allocH(128 * 128);
  h16* kW16    = allocH(128 * 128);
  h16* vW16    = allocH(128 * 128);
  h16* gcWih16 = allocH(384 * 128);
  h16* gcWhh16 = allocH(384 * 128);
  h16* mcWih16 = allocH(384 * 128);
  h16* mcWhh16 = allocH(384 * 128);
  h16* mlpW116 = allocH(256 * 256);
  h16* mlpW216 = allocH(128 * 256);
  h16* ppW116  = allocH(512 * 256);
  h16* ppW216  = allocH(32 * 512);
  h16* hid16   = allocH((size_t)BN * 256);
  h16* m16     = allocH((size_t)BN * HH);
  h16* x16     = allocH((size_t)BN * AA);
  h16* hc16    = allocH((size_t)BN * 256);
  h16* t1_16   = allocH((size_t)BN * 256);
  h16* u1_16   = allocH((size_t)BN * 512);
  h16* wh16    = allocH((size_t)BN * 256);

  float* GIa   = allocF((size_t)BN * 384);   // reused: q/k/v, gh, t1
  float* GIb   = allocF((size_t)BN * 384);   // reused: gi, t2/hp
  float* Lf    = allocF((size_t)NM1 * BN * 2);
  float* hw    = allocF((size_t)BN * NM1);
  float* u1    = allocF((size_t)BN * 512);
  float* u2    = allocF((size_t)BN * 32);
  float* stats = allocF(2 * 512);

  float* qb = GIa;
  float* kb = GIa + (size_t)BN * 128;
  float* vb = GIa + (size_t)BN * 256;
  float* gi = GIb;
  float* gh = GIa;
  float* t1 = GIa;
  float* t2 = GIb;

  auto g256 = [](int total) { return (total + 255) / 256; };
  auto cvt = [&](const float* src, int srcLd, int srcOff, int srcCols, h16* dst, int dstLd, int rows) {
    cvt16_kernel<<<g256(rows * dstLd), 256, 0, stream>>>(src, srcLd, srcOff, srcCols, dst, dstLd, rows);
  };
  // 32x64 per-wave tile for N%64==0, 32x32 for the final N=32 GEMM
  auto gemm = [&](const h16* A, int lda, const h16* W, int ldw, const float* bias,
                  float* C, h16* C16, int M, int N, int K, int act) {
    if (N % 64 == 0) {
      int tiles = (M / 32) * (N / 64);
      gemm16_kernel<2, 4><<<(tiles + 3) / 4, 128, 0, stream>>>(A, lda, W, ldw, bias, C, C16, M, N, K, act);
    } else {
      int tiles = (M / 32) * (N / 32);
      gemm16_kernel<2, 2><<<(tiles + 3) / 4, 128, 0, stream>>>(A, lda, W, ldw, bias, C, C16, M, N, K, act);
    }
  };

  // ---- weight / activation conversions to f16 ----
  cvt(P(P_ENC_W), 59, 0, 59, encW16, 64, 128);
  cvt(P(P_GF_WIH), 256, 0, 128, gfA16, 128, 384);
  cvt(P(P_GF_WIH), 256, 128, 128, gfB16, 128, 384);
  cvt(P(P_GB_WIH), 256, 0, 128, gbA16, 128, 384);
  cvt(P(P_GB_WIH), 256, 128, 128, gbB16, 128, 384);
  cvt(P(P_GF_WHH), 128, 0, 128, gfWhh16, 128, 384);
  cvt(P(P_GB_WHH), 128, 0, 128, gbWhh16, 128, 384);
  cvt(P(P_Q_W), 128, 0, 128, qW16, 128, 128);
  cvt(P(P_K_W), 128, 0, 128, kW16, 128, 128);
  cvt(P(P_V_W), 128, 0, 128, vW16, 128, 128);
  cvt(P(P_GC_WIH), 128, 0, 128, gcWih16, 128, 384);
  cvt(P(P_GC_WHH), 128, 0, 128, gcWhh16, 128, 384);
  cvt(P(P_MC_WIH), 128, 0, 128, mcWih16, 128, 384);
  cvt(P(P_MC_WHH), 128, 0, 128, mcWhh16, 128, 384);
  cvt(P(P_MLP_W1), 256, 0, 256, mlpW116, 256, 256);
  cvt(P(P_MLP_W2), 256, 0, 256, mlpW216, 256, 128);
  cvt(P(P_PP_W1), 256, 0, 256, ppW116, 256, 512);
  cvt(P(P_PP_W2), 512, 0, 512, ppW216, 512, 32);
  cvt(hidden, 256, 0, 256, hid16, 256, BN);

  // ---- encoder ----
  build_gat_kernel<<<g256(BN * 64), 256, 0, stream>>>(inp, Xgat);
  gemm(Xgat, 64, encW16, 64, P(P_ENC_B), nullptr, hEnc16, BN, 128, 64, 1);

  // ---- forward GRU: factored input projection + persistent recurrence ----
  gemm(hEnc16, 128, gfA16, 128, P(P_GF_BIH), GIa, nullptr, BN, 384, 128, 0);
  gemm(hEnc16, 128, gfB16, 128, nullptr,     GIb, nullptr, BN, 384, 128, 0);
  gru_seq_kernel<<<BN / 16, 128, GRU_SMEM, stream>>>(GIa, GIb, P(P_GF_BHH), gfWhh16,
                                                     P(P_HARD_W), P(P_HARD_B), Lf, nullptr, nullptr, 0);

  // ---- backward GRU (reuses GIa/GIb) + fused Gumbel hard gate -> hw ----
  gemm(hEnc16, 128, gbA16, 128, P(P_GB_BIH), GIa, nullptr, BN, 384, 128, 0);
  gemm(hEnc16, 128, gbB16, 128, nullptr,     GIb, nullptr, BN, 384, 128, 0);
  gru_seq_kernel<<<BN / 16, 128, GRU_SMEM, stream>>>(GIa, GIb, P(P_GB_BHH), gbWhh16,
                                                     P(P_HARD_W), P(P_HARD_B), nullptr, Lf, hw, 1);

  // ---- attention ----
  gemm(hEnc16, 128, qW16, 128, nullptr,  qb, nullptr, BN, 128, 128, 0);
  gemm(hEnc16, 128, kW16, 128, nullptr,  kb, nullptr, BN, 128, 128, 0);
  gemm(hEnc16, 128, vW16, 128, P(P_V_B), vb, nullptr, BN, 128, 128, 1);
  attention_kernel<<<BN, 128, 0, stream>>>(qb, kb, vb, hw, x16);

  // ---- GAT GRU cell ----
  gemm(x16, 128,  gcWih16, 128, P(P_GC_BIH), gi, nullptr, BN, 384, 128, 0);
  gemm(hid16, 256, gcWhh16, 128, P(P_GC_BHH), gh, nullptr, BN, 384, 128, 0);
  grucell_kernel<<<g256(BN * HH), 256, 0, stream>>>(gi, gh, hidden, 0, out_hc, hc16, 0);

  // ---- motion GRU cell ----
  motion_kernel<<<g256(BN * HH), 256, 0, stream>>>(inp, P(P_MOT_W), P(P_MOT_B), m16);
  gemm(m16, 128, mcWih16, 128, P(P_MC_BIH), gi, nullptr, BN, 384, 128, 0);
  gemm(hid16 + 128, 256, mcWhh16, 128, P(P_MC_BHH), gh, nullptr, BN, 384, 128, 0);
  grucell_kernel<<<g256(BN * HH), 256, 0, stream>>>(gi, gh, hidden, 128, out_hc, hc16, 128);

  // ---- MLP with BatchNorm ----
  gemm(hc16, 256, mlpW116, 256, P(P_MLP_B1), t1, nullptr, BN, 256, 256, 0);
  colstats_kernel<<<256, 256, 0, stream>>>(t1, BN, 256, stats);
  bnrelu_kernel<<<g256(BN * 256), 256, 0, stream>>>(t1, stats, P(P_MLP_G1), P(P_MLP_BE1),
                                                    BN, 256, nullptr, t1_16);
  gemm(t1_16, 256, mlpW216, 256, P(P_MLP_B2), t2, nullptr, BN, 128, 256, 0);
  colstats_kernel<<<128, 256, 0, stream>>>(t2, BN, 128, stats);
  bnrelu_kernel<<<g256(BN * 128), 256, 0, stream>>>(t2, stats, P(P_MLP_G2), P(P_MLP_BE2),
                                                    BN, 128, t2, nullptr);

  // ---- wh = 0.05 * [pe, hp]  (at-branch is dead: softmax over a length-1 axis) ----
  buildwh_kernel<<<g256(BN * 256), 256, 0, stream>>>(inp, P(P_SPAT_W), P(P_SPAT_B), t2, wh16);

  // ---- pool projector ----
  gemm(wh16, 256, ppW116, 256, P(P_PP_B1), u1, nullptr, BN, 512, 256, 0);
  colstats_kernel<<<512, 256, 0, stream>>>(u1, BN, 512, stats);
  bnrelu_kernel<<<g256(BN * 512), 256, 0, stream>>>(u1, stats, P(P_PP_G1), P(P_PP_BE1),
                                                    BN, 512, nullptr, u1_16);
  gemm(u1_16, 512, ppW216, 512, P(P_PP_B2), u2, nullptr, BN, 32, 512, 0);
  colstats_kernel<<<32, 256, 0, stream>>>(u2, BN, 32, stats);
  bnrelu_kernel<<<g256(BN * 32), 256, 0, stream>>>(u2, stats, P(P_PP_G2), P(P_PP_BE2),
                                                   BN, 32, out_pool, nullptr);
}
